// LinearAttention_24850680775182
// MI455X (gfx1250) — compile-verified, matrix operand order checked by reading, not run on hardware
//
#include <hip/hip_runtime.h>

typedef __bf16 bf16_t;
typedef __attribute__((ext_vector_type(16))) __bf16 v16bf;
typedef __attribute__((ext_vector_type(8)))  __bf16 v8bf;
typedef __attribute__((ext_vector_type(8)))  float  v8f;

#define D_MODEL 1024
#define NHEAD   16
#define HDIM    64
#define BATCH   4
#define SEQ     4096
#define ROWS    (BATCH*SEQ)      // 16384
#define NHTOT   (BATCH*NHEAD)    // 64 (b*16+h)

#define LDS_STRIDE 36            // 32 + 4 pad (bf16 elems) to spread ds banks

// ---------------------------------------------------------------------------
// WMMA helpers
// ---------------------------------------------------------------------------
__device__ __forceinline__ v8f wmma_bf16(v16bf a, v16bf b, v8f c) {
  // D = A(16x32) * B(32x16) + C, f32 accumulate
  return __builtin_amdgcn_wmma_f32_16x16x32_bf16(false, a, false, b, (short)0, c,
                                                 false, false);
}

// Load a 16x32 bf16 fragment in the ISA A-matrix layout (also used for B^T,
// i.e. W rows). Per ISA 7.12.2: lane = hi*16 + row; VGPR0..3 hold
// K = hi*8 + 0..7, VGPR4..7 hold K = 16 + hi*8 + 0..7 -> two contiguous
// 16-byte chunks per lane.
__device__ __forceinline__ v16bf load_frag(const bf16_t* base, int ld) {
  int lane = threadIdx.x & 31;
  int row  = lane & 15;
  int hi   = lane >> 4;
  const bf16_t* p = base + (size_t)row * ld + hi * 8;
  v8bf lo = *(const v8bf*)(p);
  v8bf hh = *(const v8bf*)(p + 16);
  v16bf r;
#pragma unroll
  for (int i = 0; i < 8; ++i) { r[i] = lo[i]; r[i + 8] = hh[i]; }
  return r;
}

__device__ __forceinline__ float elu1(float f) {
  return f > 0.0f ? f + 1.0f : __expf(f);
}

// ---------------------------------------------------------------------------
// CDNA5 async global->LDS copy (ASYNCcnt-tracked, bypasses VGPRs).
// vdst = per-lane LDS byte address (low 32 bits of the flat LDS pointer),
// vaddr = per-lane 64-bit global address (GV mode, saddr = off).
// ---------------------------------------------------------------------------
__device__ __forceinline__ void async_copy_b128(unsigned lds_byte_off,
                                                const bf16_t* g) {
  asm volatile("global_load_async_to_lds_b128 %0, %1, off"
               :: "v"(lds_byte_off), "v"(g)
               : "memory");
}
__device__ __forceinline__ void wait_async_zero() {
  asm volatile("s_wait_asynccnt 0x0" ::: "memory");
}
__device__ __forceinline__ unsigned lds_off32(const void* p) {
  // LDS aperture occupies addr[63:32]; addr[31:0] is the LDS byte offset.
  return (unsigned)(size_t)p;
}

// ---------------------------------------------------------------------------
// fp32 -> bf16 conversion
// ---------------------------------------------------------------------------
__global__ void cvt_f32_bf16(const float* __restrict__ in, bf16_t* __restrict__ out, int n) {
  int i = (blockIdx.x * blockDim.x + threadIdx.x) * 4;
  if (i + 3 < n) {
    float4 f = *(const float4*)(in + i);
    out[i + 0] = (bf16_t)f.x;
    out[i + 1] = (bf16_t)f.y;
    out[i + 2] = (bf16_t)f.z;
    out[i + 3] = (bf16_t)f.w;
  } else {
    for (int j = i; j < n; ++j) out[j] = (bf16_t)in[j];
  }
}

// ---------------------------------------------------------------------------
// Shared GEMM core: C[m][n] = sum_k A[m][k] * W[n][k], bf16 in, f32 acc.
// Block tile 128(M) x 128(N), K stepped by 32 through double-buffered LDS
// filled with global_load_async_to_lds_b128. 256 threads = 8 waves as
// 2(M) x 4(N); each wave computes 64x32 (4x2 16x16 tiles): 8 wmma / k-step.
// Pipeline per iter: s_wait_asynccnt 0 -> barrier -> issue next tile async
// -> ds_load fragments -> 8x wmma.
// ---------------------------------------------------------------------------
#define GEMM_CORE(Aptr, Wptr, Kdim)                                            \
  __shared__ bf16_t sA[2 * 128 * LDS_STRIDE];                                  \
  __shared__ bf16_t sW[2 * 128 * LDS_STRIDE];                                  \
  const int tid  = threadIdx.x;                                                \
  const int lrow = tid >> 1;              /* 0..127 */                         \
  const int lcol = (tid & 1) * 16;        /* 0 or 16 */                        \
  const bf16_t* gA = (Aptr) + (size_t)(tile_m + lrow) * (Kdim) + lcol;         \
  const bf16_t* gW = (Wptr) + (size_t)(tile_n + lrow) * (Kdim) + lcol;         \
  const unsigned aoff = lds_off32(sA + lrow * LDS_STRIDE + lcol);              \
  const unsigned woff = lds_off32(sW + lrow * LDS_STRIDE + lcol);              \
  const unsigned BUFB = 128 * LDS_STRIDE * 2;  /* bytes per buffer */          \
  const int wave  = tid >> 5;                                                  \
  const int waveM = wave & 1;             /* 0..1 */                           \
  const int waveN = wave >> 1;            /* 0..3 */                           \
  const int wm0 = waveM * 64;                                                  \
  const int wn0 = waveN * 32;                                                  \
  v8f acc[4][2] = {};                                                          \
  async_copy_b128(aoff,      gA);                                              \
  async_copy_b128(aoff + 16, gA + 8);                                          \
  async_copy_b128(woff,      gW);                                              \
  async_copy_b128(woff + 16, gW + 8);                                          \
  for (int k = 0; k < (Kdim); k += 32) {                                       \
    const int cur = (k >> 5) & 1;                                              \
    wait_async_zero();                                                         \
    __syncthreads();                                                           \
    if (k + 32 < (Kdim)) {                                                     \
      const unsigned nb = (unsigned)(1 - cur) * BUFB;                          \
      async_copy_b128(aoff + nb,      gA + k + 32);                            \
      async_copy_b128(aoff + nb + 16, gA + k + 40);                            \
      async_copy_b128(woff + nb,      gW + k + 32);                            \
      async_copy_b128(woff + nb + 16, gW + k + 40);                            \
    }                                                                          \
    const bf16_t* bufA = sA + cur * 128 * LDS_STRIDE;                          \
    const bf16_t* bufW = sW + cur * 128 * LDS_STRIDE;                          \
    v16bf af[4];                                                               \
    _Pragma("unroll")                                                          \
    for (int mt = 0; mt < 4; ++mt)                                             \
      af[mt] = load_frag(bufA + (wm0 + mt * 16) * LDS_STRIDE, LDS_STRIDE);     \
    v16bf wf[2];                                                               \
    _Pragma("unroll")                                                          \
    for (int nt = 0; nt < 2; ++nt)                                             \
      wf[nt] = load_frag(bufW + (wn0 + nt * 16) * LDS_STRIDE, LDS_STRIDE);     \
    _Pragma("unroll")                                                          \
    for (int mt = 0; mt < 4; ++mt)                                             \
      _Pragma("unroll")                                                        \
      for (int nt = 0; nt < 2; ++nt)                                           \
        acc[mt][nt] = wmma_bf16(af[mt], wf[nt], acc[mt][nt]);                  \
  }                                                                            \
  const int lane = threadIdx.x & 31;                                           \
  const int ncol = lane & 15;                                                  \
  const int mhi  = (lane >> 4) * 8;

// ---------------------------------------------------------------------------
// Kernel 1: qkv = x @ Wqkv^T ; fuse elu+1 on q,k ; scatter q (merged rows),
// kT and vT (per-head transposed: [head][d][t]).
// ---------------------------------------------------------------------------
__global__ __launch_bounds__(256) void gemm_qkv(
    const bf16_t* __restrict__ X,     // [16384,1024]
    const bf16_t* __restrict__ W,     // [3072,1024]
    bf16_t* __restrict__ qbuf,        // [16384,1024]
    bf16_t* __restrict__ kT,          // [64,64,4096]
    bf16_t* __restrict__ vT)          // [64,64,4096]
{
  const int tile_n = blockIdx.x * 128;
  const int tile_m = blockIdx.y * 128;
  GEMM_CORE(X, W, D_MODEL)

#pragma unroll
  for (int mt = 0; mt < 4; ++mt)
#pragma unroll
    for (int nt = 0; nt < 2; ++nt)
#pragma unroll
      for (int i = 0; i < 8; ++i) {
        int r = tile_m + wm0 + mt * 16 + mhi + i;  // global row (b*4096+t)
        int c = tile_n + wn0 + nt * 16 + ncol;     // col in [0,3072)
        float f = acc[mt][nt][i];
        int b = r >> 12, t = r & 4095;
        if (c < D_MODEL) {                         // q: elu+1, merged layout
          qbuf[(size_t)r * D_MODEL + c] = (bf16_t)elu1(f);
        } else if (c < 2 * D_MODEL) {              // k: elu+1, transposed per head
          int hd = c - D_MODEL;
          int h = hd >> 6, d = hd & 63;
          kT[(((size_t)(b * NHEAD + h)) * HDIM + d) * SEQ + t] = (bf16_t)elu1(f);
        } else {                                   // v: raw, transposed per head
          int hd = c - 2 * D_MODEL;
          int h = hd >> 6, d = hd & 63;
          vT[(((size_t)(b * NHEAD + h)) * HDIM + d) * SEQ + t] = (bf16_t)f;
        }
      }
}

// ---------------------------------------------------------------------------
// Kernel 2: per head: kv[d][e] = sum_t k[t,d] v[t,e]  (WMMA over K=4096)
// stored transposed+bf16 as kvT[head][e][d]; also k_sum[head][d].
// 4 waves: each wave owns a 16-row d-strip x full 64-wide e.
// ---------------------------------------------------------------------------
__global__ __launch_bounds__(128) void kv_kernel(
    const bf16_t* __restrict__ kT,    // [64,64,4096]
    const bf16_t* __restrict__ vT,    // [64,64,4096]
    bf16_t* __restrict__ kvT,         // [64,64,64]  ([head][e][d])
    float* __restrict__ ksum)         // [64,64]
{
  int head = blockIdx.x;
  const bf16_t* Kh = kT + (size_t)head * HDIM * SEQ;
  const bf16_t* Vh = vT + (size_t)head * HDIM * SEQ;
  int wave = threadIdx.x >> 5;
  int d0 = wave * 16;

  v8f acc[4] = {};
  for (int t = 0; t < SEQ; t += 32) {
    v16bf a = load_frag(Kh + (size_t)d0 * SEQ + t, SEQ);
#pragma unroll
    for (int nt = 0; nt < 4; ++nt) {
      v16bf b = load_frag(Vh + (size_t)(nt * 16) * SEQ + t, SEQ);
      acc[nt] = wmma_bf16(a, b, acc[nt]);
    }
  }

  int lane = threadIdx.x & 31;
  int e_l  = lane & 15;
  int mhi  = (lane >> 4) * 8;
#pragma unroll
  for (int nt = 0; nt < 4; ++nt)
#pragma unroll
    for (int i = 0; i < 8; ++i) {
      int d = d0 + mhi + i;
      int e = nt * 16 + e_l;
      kvT[(size_t)head * HDIM * HDIM + (size_t)e * HDIM + d] = (bf16_t)acc[nt][i];
    }

  // k_sum[d] = sum_t k[t,d] = row-sum of kT
  __shared__ float s[HDIM];
  if (threadIdx.x < HDIM) s[threadIdx.x] = 0.0f;
  __syncthreads();
  int d    = threadIdx.x & 63;
  int half = threadIdx.x >> 6;
  const bf16_t* row = Kh + (size_t)d * SEQ + half * (SEQ / 2);
  float part = 0.0f;
  for (int t = 0; t < SEQ / 2; ++t) part += (float)row[t];
  atomicAdd(&s[d], part);
  __syncthreads();
  if (threadIdx.x < HDIM) ksum[head * HDIM + threadIdx.x] = s[threadIdx.x];
}

// ---------------------------------------------------------------------------
// Kernel 3: per (head, 64-row tile): out = (q @ kv) * z, merged-head bf16.
// z = 1/(q . k_sum + eps) computed into LDS first. 4 waves, 16 rows each.
// ---------------------------------------------------------------------------
__global__ __launch_bounds__(128) void attn_kernel(
    const bf16_t* __restrict__ qbuf,  // [16384,1024]
    const bf16_t* __restrict__ kvT,   // [64,64,64] ([head][e][d])
    const float* __restrict__ ksum,   // [64,64]
    bf16_t* __restrict__ attn)        // [16384,1024]
{
  int head = blockIdx.y;              // b*16 + h
  int b = head >> 4, h = head & 15;
  size_t rowBase = (size_t)b * SEQ + (size_t)blockIdx.x * 64;  // into 16384 rows
  const bf16_t* Q   = qbuf + rowBase * D_MODEL + h * HDIM;
  const bf16_t* kvh = kvT + (size_t)head * HDIM * HDIM;

  __shared__ float ks[HDIM];
  __shared__ float zs[64];
  if (threadIdx.x < HDIM) ks[threadIdx.x] = ksum[head * HDIM + threadIdx.x];
  __syncthreads();
  if (threadIdx.x < 64) {
    const bf16_t* qr = Q + (size_t)threadIdx.x * D_MODEL;
    float dot = 0.0f;
#pragma unroll
    for (int d = 0; d < HDIM; ++d) dot += (float)qr[d] * ks[d];
    zs[threadIdx.x] = 1.0f / (dot + 1e-6f);
  }
  __syncthreads();

  int wave = threadIdx.x >> 5;
  int m0 = wave * 16;
  v8f acc[4] = {};
#pragma unroll
  for (int k = 0; k < HDIM; k += 32) {
    v16bf a = load_frag(Q + (size_t)m0 * D_MODEL + k, D_MODEL);
#pragma unroll
    for (int nt = 0; nt < 4; ++nt) {
      v16bf w = load_frag(kvh + (size_t)(nt * 16) * HDIM + k, HDIM);
      acc[nt] = wmma_bf16(a, w, acc[nt]);
    }
  }

  int lane = threadIdx.x & 31;
  int e_l  = lane & 15;
  int mhi  = (lane >> 4) * 8;
#pragma unroll
  for (int nt = 0; nt < 4; ++nt)
#pragma unroll
    for (int i = 0; i < 8; ++i) {
      int lm = m0 + mhi + i;             // local row 0..63
      int e  = nt * 16 + e_l;
      float f = acc[nt][i] * zs[lm];
      attn[(rowBase + lm) * D_MODEL + h * HDIM + e] = (bf16_t)f;
    }
}

// ---------------------------------------------------------------------------
// Kernel 4: out = attn @ Wout^T + bout   (fp32 output)
// ---------------------------------------------------------------------------
__global__ __launch_bounds__(256) void gemm_out(
    const bf16_t* __restrict__ A,     // [16384,1024]
    const bf16_t* __restrict__ W,     // [1024,1024]
    const float* __restrict__ bias,   // [1024]
    float* __restrict__ out)          // [16384,1024]
{
  const int tile_n = blockIdx.x * 128;
  const int tile_m = blockIdx.y * 128;
  GEMM_CORE(A, W, D_MODEL)

#pragma unroll
  for (int mt = 0; mt < 4; ++mt)
#pragma unroll
    for (int nt = 0; nt < 2; ++nt)
#pragma unroll
      for (int i = 0; i < 8; ++i) {
        int r = tile_m + wm0 + mt * 16 + mhi + i;
        int c = tile_n + wn0 + nt * 16 + ncol;
        out[(size_t)r * D_MODEL + c] = acc[mt][nt][i] + bias[c];
      }
}

// ---------------------------------------------------------------------------
// Host launcher
// ---------------------------------------------------------------------------
extern "C" void kernel_launch(void* const* d_in, const int* in_sizes, int n_in,
                              void* d_out, int out_size, void* d_ws, size_t ws_size,
                              hipStream_t stream) {
  const float* x    = (const float*)d_in[0];
  const float* Wqkv = (const float*)d_in[1];
  const float* Wout = (const float*)d_in[2];
  const float* bout = (const float*)d_in[3];
  float* out = (float*)d_out;

  char* ws = (char*)d_ws;
  size_t off = 0;
  auto take = [&](size_t nbytes) -> char* {
    char* p = ws + off;
    off += (nbytes + 255) & ~(size_t)255;
    return p;
  };
  bf16_t* xb   = (bf16_t*)take((size_t)ROWS * D_MODEL * 2);
  bf16_t* wqb  = (bf16_t*)take((size_t)3 * D_MODEL * D_MODEL * 2);
  bf16_t* wob  = (bf16_t*)take((size_t)D_MODEL * D_MODEL * 2);
  bf16_t* qbuf = (bf16_t*)take((size_t)ROWS * D_MODEL * 2);
  bf16_t* kT   = (bf16_t*)take((size_t)NHTOT * HDIM * SEQ * 2);
  bf16_t* vT   = (bf16_t*)take((size_t)NHTOT * HDIM * SEQ * 2);
  bf16_t* kvb  = (bf16_t*)take((size_t)NHTOT * HDIM * HDIM * 2);
  float*  ksum = (float*)take((size_t)NHTOT * HDIM * 4);
  bf16_t* attn = (bf16_t*)take((size_t)ROWS * D_MODEL * 2);

  int n;
  n = ROWS * D_MODEL;
  cvt_f32_bf16<<<(n / 4 + 255) / 256, 256, 0, stream>>>(x, xb, n);
  n = 3 * D_MODEL * D_MODEL;
  cvt_f32_bf16<<<(n / 4 + 255) / 256, 256, 0, stream>>>(Wqkv, wqb, n);
  n = D_MODEL * D_MODEL;
  cvt_f32_bf16<<<(n / 4 + 255) / 256, 256, 0, stream>>>(Wout, wob, n);

  gemm_qkv<<<dim3(3 * D_MODEL / 128, ROWS / 128), 256, 0, stream>>>(xb, wqb, qbuf, kT, vT);
  kv_kernel<<<NHTOT, 128, 0, stream>>>(kT, vT, kvb, ksum);
  attn_kernel<<<dim3(SEQ / 64, NHTOT), 128, 0, stream>>>(qbuf, kvb, ksum, attn);
  gemm_out<<<dim3(D_MODEL / 128, ROWS / 128), 256, 0, stream>>>(attn, wob, bout, out);
}
